// GAT_48172353192429
// MI455X (gfx1250) — matrix operand order repaired and checked by reading, not compiled, not run on hardware
//
#include <hip/hip_runtime.h>
#include <hip/hip_bf16.h>

// ---------------- problem constants (match reference) ----------------
#define N_NODES 20000
#define N_EDGES 320000
#define N_GRAPH 32
#define N_HEAD  8
#define N_CH    56
#define HC      448          // N_HEAD * N_CH
#define DIN     32
#define DE      5
#define NOUT    18
#define E2      (N_EDGES + N_NODES)   // edges + self loops = 340000
#define NEG_SLOPE 0.2f
#define ENC_NEG_INF 0x007FFFFFu       // order-encoded -inf

typedef float  v8f   __attribute__((ext_vector_type(8)));
typedef __bf16 v16bf __attribute__((ext_vector_type(16)));

// ---------------- small fills ----------------
__global__ void gat_fill_f32(float* p, float v, int n) {
  int i = blockIdx.x * blockDim.x + threadIdx.x;
  if (i < n) p[i] = v;
}
__global__ void gat_fill_u32(unsigned* p, unsigned v, int n) {
  int i = blockIdx.x * blockDim.x + threadIdx.x;
  if (i < n) p[i] = v;
}

// ---------------- self-loop attr = mean of incoming edge_attr ----------------
__global__ void gat_selfloop_accum(const int* __restrict__ ei, const float* __restrict__ ea,
                                   float* __restrict__ deg, float* __restrict__ esum) {
  int e = blockIdx.x * blockDim.x + threadIdx.x;
  if (e >= N_EDGES) return;
  int d = ei[N_EDGES + e];                  // dst row of edge_index
  atomicAdd(&deg[d], 1.0f);
#pragma unroll
  for (int k = 0; k < DE; ++k) atomicAdd(&esum[d * DE + k], ea[e * DE + k]);
}

__global__ void gat_build_edges(const int* __restrict__ ei, const float* __restrict__ ea,
                                const float* __restrict__ deg, const float* __restrict__ esum,
                                int* __restrict__ src2, int* __restrict__ dst2,
                                float* __restrict__ ea2) {
  int e = blockIdx.x * blockDim.x + threadIdx.x;
  if (e >= E2) return;
  if (e < N_EDGES) {
    src2[e] = ei[e];
    dst2[e] = ei[N_EDGES + e];
#pragma unroll
    for (int k = 0; k < DE; ++k) ea2[e * DE + k] = ea[e * DE + k];
  } else {
    int n = e - N_EDGES;
    src2[e] = n; dst2[e] = n;
    float dg = fmaxf(deg[n], 1.0f);
#pragma unroll
    for (int k = 0; k < DE; ++k) ea2[e * DE + k] = esum[n * DE + k] / dg;
  }
}

// ---------------- bf16 WMMA GEMM: C[M,Nc] = A[M,K] @ B[K,Nc], f32 in/out ------
// Block (256 thr = 8 waves) computes a 128x64 C macro-tile. Each wave owns a
// 16-row strip and 4 column tiles -> 4 v_wmma_f32_16x16x32_bf16 per k-step,
// sharing one A fragment. The 32x64 B k-slab is staged once per block into LDS
// as bf16, K-transposed with padded stride 40 (80B) so each lane's 16-K
// fragment is two aligned ds_load_b128 and lane->bank mapping (lid*20 mod 64)
// is conflict-free.
// Fragment layouts used here (contiguous K, not the ISA table order; A and B match):
//   A 16x32 bf16: lanes 0-15 row=lid hold K 0..15; lanes 16-31 hold
//   K 16..31.  B 32x16: lanes 0-15 col=lid hold K 0..15; lanes 16-31
//   hold K 16..31.  C/D: VGPR v -> row v + 8*half, col = lid.
#define BSTRIDE 40   // padded bf16 stride per column in LDS (80 bytes)
__global__ void gat_gemm_wmma(const float* __restrict__ A, const float* __restrict__ B,
                              float* __restrict__ Cm, int M, int K, int Nc) {
  __shared__ __bf16 ldsBt[64 * BSTRIDE];          // 5120 B, [col][k] transposed
  const int nBlocks = Nc >> 6;                    // 64-wide column blocks
  const int bm = blockIdx.x / nBlocks;
  const int bn = blockIdx.x - bm * nBlocks;
  const int nb = bn * 64;
  const int waveId = threadIdx.x >> 5;
  const int lane = threadIdx.x & 31;
  const int hf  = lane >> 4;                      // 16-lane half
  const int lid = lane & 15;
  const int rowBase = bm * 128 + waveId * 16;
  const bool active = rowBase < M;                // M % 16 == 0 always here
  const int rowA = rowBase + lid;
  const float* arow = A + (size_t)rowA * K;
  // staging assignment: thread -> B[k0 + r][nb + cst .. cst+7]
  const int r   = threadIdx.x >> 3;               // 0..31
  const int cst = (threadIdx.x & 7) * 8;          // 0,8,..,56
  v8f acc[4] = {{}, {}, {}, {}};
  for (int k0 = 0; k0 < K; k0 += 32) {
    const float* brow = B + (size_t)(k0 + r) * Nc + nb + cst;
    float4 f0 = *(const float4*)brow;             // global_load_b128 x2
    float4 f1 = *(const float4*)(brow + 4);
    __syncthreads();                              // previous iter's LDS reads done
    ldsBt[(cst + 0) * BSTRIDE + r] = (__bf16)f0.x;
    ldsBt[(cst + 1) * BSTRIDE + r] = (__bf16)f0.y;
    ldsBt[(cst + 2) * BSTRIDE + r] = (__bf16)f0.z;
    ldsBt[(cst + 3) * BSTRIDE + r] = (__bf16)f0.w;
    ldsBt[(cst + 4) * BSTRIDE + r] = (__bf16)f1.x;
    ldsBt[(cst + 5) * BSTRIDE + r] = (__bf16)f1.y;
    ldsBt[(cst + 6) * BSTRIDE + r] = (__bf16)f1.z;
    ldsBt[(cst + 7) * BSTRIDE + r] = (__bf16)f1.w;
    __syncthreads();                              // slab visible to all waves
    if (active) {
      const float* ap = arow + k0 + hf * 16;
      __builtin_prefetch(ap + 32, 0, 1);          // global_prefetch_b8 next slab
      v16bf a;
#pragma unroll
      for (int i = 0; i < 8; ++i) {
        a[i]     = (__bf16)ap[i];
        a[8 + i] = (__bf16)ap[8 + i];
      }
#pragma unroll
      for (int j = 0; j < 4; ++j) {
        const uint4* q = (const uint4*)&ldsBt[(j * 16 + lid) * BSTRIDE + hf * 16];
        union { uint4 u[2]; v16bf v; } bb;
        bb.u[0] = q[0];                           // ds_load_b128 x2 (16B aligned)
        bb.u[1] = q[1];
        acc[j] = __builtin_amdgcn_wmma_f32_16x16x32_bf16(false, a, false, bb.v,
                                                         (short)0, acc[j], false, false);
      }
    }
  }
  if (active) {
#pragma unroll
    for (int j = 0; j < 4; ++j)
#pragma unroll
      for (int v = 0; v < 8; ++v)
        Cm[(size_t)(rowBase + hf * 8 + v) * Nc + nb + j * 16 + lid] = acc[j][v];
  }
}

// ---------------- edge pass 1: scores + segment max ----------------
// one thread per (edge, head); We (9KB) and att (1.8KB) staged in LDS so the
// 610MB ee tensor is never materialized.
__global__ void gat_edge_score(const float* __restrict__ xl, const float* __restrict__ xr,
                               const int* __restrict__ src2, const int* __restrict__ dst2,
                               const float* __restrict__ ea2,
                               const float* __restrict__ We, const float* __restrict__ att,
                               float* __restrict__ score, unsigned* __restrict__ mkey) {
  __shared__ float sWe[DE * HC];
  __shared__ float sAtt[HC];
  for (int i = threadIdx.x; i < DE * HC; i += blockDim.x) sWe[i] = We[i];
  for (int i = threadIdx.x; i < HC;      i += blockDim.x) sAtt[i] = att[i];
  __syncthreads();
  int t = blockIdx.x * blockDim.x + threadIdx.x;
  if (t >= E2 * N_HEAD) return;
  int e = t >> 3, h = t & 7;
  int s = src2[e], d = dst2[e];
  float eav[DE];
#pragma unroll
  for (int k = 0; k < DE; ++k) eav[k] = ea2[e * DE + k];
  const float* xls = xl + (size_t)s * HC;
  const float* xrd = xr + (size_t)d * HC;
  float acc = 0.0f;
  const int base = h * N_CH;
#pragma unroll 4
  for (int c = 0; c < N_CH; ++c) {
    int ch = base + c;
    float ee = eav[0] * sWe[ch] + eav[1] * sWe[HC + ch] + eav[2] * sWe[2 * HC + ch]
             + eav[3] * sWe[3 * HC + ch] + eav[4] * sWe[4 * HC + ch];
    float v = xls[ch] + xrd[ch] + ee;
    v = (v > 0.0f) ? v : v * NEG_SLOPE;          // leaky relu
    acc += v * sAtt[ch];
  }
  score[t] = acc;
  unsigned u = __float_as_uint(acc);             // order-preserving encode
  unsigned key = (u & 0x80000000u) ? ~u : (u | 0x80000000u);
  atomicMax(&mkey[d * N_HEAD + h], key);
}

// ---------------- edge pass 2: exp(score - max) + segment sum ----------------
__global__ void gat_edge_expsum(const int* __restrict__ dst2,
                                const unsigned* __restrict__ mkey,
                                float* __restrict__ score, float* __restrict__ ssum) {
  int t = blockIdx.x * blockDim.x + threadIdx.x;
  if (t >= E2 * N_HEAD) return;
  int e = t >> 3, h = t & 7;
  int d = dst2[e];
  unsigned key = mkey[d * N_HEAD + h];
  unsigned u = (key & 0x80000000u) ? (key & 0x7FFFFFFFu) : ~key;
  float m = __uint_as_float(u);
  float w = expf(score[t] - m);
  score[t] = w;
  atomicAdd(&ssum[d * N_HEAD + h], w);
}

// ---------------- edge pass 3: out[dst] += alpha * xl[src] ----------------
__global__ void gat_edge_aggregate(const float* __restrict__ xl,
                                   const int* __restrict__ src2, const int* __restrict__ dst2,
                                   const float* __restrict__ score, const float* __restrict__ ssum,
                                   float* __restrict__ agg) {
  int t = blockIdx.x * blockDim.x + threadIdx.x;
  if (t >= E2 * N_HEAD) return;
  int e = t >> 3, h = t & 7;
  int s = src2[e], d = dst2[e];
  float alpha = score[t] / (ssum[d * N_HEAD + h] + 1e-16f);
  const float* xls = xl + (size_t)s * HC + h * N_CH;
  float* od = agg + (size_t)d * HC + h * N_CH;
#pragma unroll 4
  for (int c = 0; c < N_CH; ++c) atomicAdd(&od[c], alpha * xls[c]);
}

// ---------------- bias + ELU (in place) ----------------
__global__ void gat_bias_elu(float* __restrict__ agg, const float* __restrict__ bias) {
  int i = blockIdx.x * blockDim.x + threadIdx.x;
  if (i >= N_NODES * HC) return;
  int ch = i % HC;
  float v = agg[i] + bias[ch];
  agg[i] = (v > 0.0f) ? v : (expf(v) - 1.0f);
}

// ---------------- BN stats: one block per channel, no atomics ----------------
__global__ void gat_bn_stats(const float* __restrict__ x, float* __restrict__ meanOut,
                             float* __restrict__ rstdOut, int rows) {
  __shared__ float r1[256], r2[256];
  int ch = blockIdx.x;
  float s = 0.0f, s2 = 0.0f;
  for (int n = threadIdx.x; n < rows; n += blockDim.x) {
    float v = x[(size_t)n * HC + ch];
    s += v; s2 += v * v;
  }
  r1[threadIdx.x] = s; r2[threadIdx.x] = s2;
  __syncthreads();
  for (int k = blockDim.x >> 1; k > 0; k >>= 1) {
    if ((int)threadIdx.x < k) { r1[threadIdx.x] += r1[threadIdx.x + k];
                                r2[threadIdx.x] += r2[threadIdx.x + k]; }
    __syncthreads();
  }
  if (threadIdx.x == 0) {
    float mean = r1[0] / rows;
    float var  = r2[0] / rows - mean * mean;   // biased, like torch BN training
    meanOut[ch] = mean;
    rstdOut[ch] = rsqrtf(var + 1e-5f);
  }
}

__global__ void gat_bn_apply(const float* __restrict__ x, const float* __restrict__ mean,
                             const float* __restrict__ rstd, const float* __restrict__ gamma,
                             const float* __restrict__ beta, float* __restrict__ y) {
  int i = blockIdx.x * blockDim.x + threadIdx.x;
  if (i >= N_NODES * HC) return;
  int ch = i % HC;
  y[i] = (x[i] - mean[ch]) * rstd[ch] * gamma[ch] + beta[ch];
}

// ---------------- global mean pool (bias2 folded in) ----------------
__global__ void gat_pool(const float* __restrict__ agg, const float* __restrict__ bias2,
                         const int* __restrict__ batch, float* __restrict__ pooled,
                         float* __restrict__ cnt) {
  int i = blockIdx.x * blockDim.x + threadIdx.x;
  if (i >= N_NODES * HC) return;
  int n = i / HC, ch = i - n * HC;
  int g = batch[n];
  atomicAdd(&pooled[g * HC + ch], agg[i] + bias2[ch]);
  if (ch == 0) atomicAdd(&cnt[g], 1.0f);
}

// ---------------- single-block head: mean, BN over G, linear, log_softmax ----
__global__ void gat_final(const float* __restrict__ pooled, const float* __restrict__ cnt,
                          const float* __restrict__ gamma, const float* __restrict__ beta,
                          const float* __restrict__ Wlin, const float* __restrict__ blin,
                          float* __restrict__ out) {
  __shared__ float pm[N_GRAPH * HC];      // 57344 B
  __shared__ float lg[N_GRAPH * NOUT];
  __shared__ float rmax[N_GRAPH], rlse[N_GRAPH];
  int tid = threadIdx.x;
  for (int i = tid; i < N_GRAPH * HC; i += blockDim.x) {
    int g = i / HC;
    float c = fmaxf(cnt[g], 1.0f);
    pm[i] = pooled[i] / c;
  }
  __syncthreads();
  for (int ch = tid; ch < HC; ch += blockDim.x) {
    float s = 0.0f, s2 = 0.0f;
    for (int g = 0; g < N_GRAPH; ++g) { float v = pm[g * HC + ch]; s += v; s2 += v * v; }
    float mean = s / N_GRAPH;
    float var  = s2 / N_GRAPH - mean * mean;
    float rstd = rsqrtf(var + 1e-5f);
    float ga = gamma[ch], be = beta[ch];
    for (int g = 0; g < N_GRAPH; ++g)
      pm[g * HC + ch] = (pm[g * HC + ch] - mean) * rstd * ga + be;
  }
  __syncthreads();
  for (int t = tid; t < N_GRAPH * NOUT; t += blockDim.x) {
    int g = t / NOUT, o = t - g * NOUT;
    float acc = blin[o];
    for (int ch = 0; ch < HC; ++ch) acc += pm[g * HC + ch] * Wlin[ch * NOUT + o];
    lg[t] = acc;
  }
  __syncthreads();
  if (tid < N_GRAPH) {
    float mx = -1e30f;
    for (int o = 0; o < NOUT; ++o) mx = fmaxf(mx, lg[tid * NOUT + o]);
    float s = 0.0f;
    for (int o = 0; o < NOUT; ++o) s += expf(lg[tid * NOUT + o] - mx);
    rmax[tid] = mx; rlse[tid] = logf(s);
  }
  __syncthreads();
  for (int t = tid; t < N_GRAPH * NOUT; t += blockDim.x) {
    int g = t / NOUT;
    out[t] = lg[t] - rmax[g] - rlse[g];
  }
}

// ---------------- host orchestration ----------------
extern "C" void kernel_launch(void* const* d_in, const int* in_sizes, int n_in,
                              void* d_out, int out_size, void* d_ws, size_t ws_size,
                              hipStream_t stream) {
  (void)in_sizes; (void)n_in; (void)out_size; (void)ws_size;
  const float* x     = (const float*)d_in[0];
  const int*   ei    = (const int*)  d_in[1];
  const float* eattr = (const float*)d_in[2];
  const int*   batch = (const int*)  d_in[3];
  const float* Wl1   = (const float*)d_in[4];
  const float* Wr1   = (const float*)d_in[5];
  const float* We1   = (const float*)d_in[6];
  const float* att1  = (const float*)d_in[7];
  const float* bias1 = (const float*)d_in[8];
  const float* Wl2   = (const float*)d_in[9];
  const float* Wr2   = (const float*)d_in[10];
  const float* We2   = (const float*)d_in[11];
  const float* att2  = (const float*)d_in[12];
  const float* bias2 = (const float*)d_in[13];
  const float* gamma = (const float*)d_in[14];
  const float* beta  = (const float*)d_in[15];
  const float* Wlin  = (const float*)d_in[16];
  const float* blin  = (const float*)d_in[17];
  float* out = (float*)d_out;

  size_t off = 0;
  auto carve = [&](size_t bytes) -> void* {
    void* p = (char*)d_ws + off;
    off += (bytes + 255) & ~(size_t)255;
    return p;
  };
  int*      src2   = (int*)     carve(sizeof(int)   * E2);
  int*      dst2   = (int*)     carve(sizeof(int)   * E2);
  float*    ea2    = (float*)   carve(sizeof(float) * (size_t)E2 * DE);
  float*    deg    = (float*)   carve(sizeof(float) * N_NODES);
  float*    esum   = (float*)   carve(sizeof(float) * N_NODES * DE);
  float*    xl     = (float*)   carve(sizeof(float) * (size_t)N_NODES * HC);
  float*    xr     = (float*)   carve(sizeof(float) * (size_t)N_NODES * HC);
  float*    agg    = (float*)   carve(sizeof(float) * (size_t)N_NODES * HC);
  float*    hbn    = (float*)   carve(sizeof(float) * (size_t)N_NODES * HC);
  float*    wbuf   = (float*)   carve(sizeof(float) * (size_t)E2 * N_HEAD);
  unsigned* mkey   = (unsigned*)carve(sizeof(unsigned) * N_NODES * N_HEAD);
  float*    ssum   = (float*)   carve(sizeof(float) * N_NODES * N_HEAD);
  float*    bn_mean= (float*)   carve(sizeof(float) * HC);
  float*    bn_rstd= (float*)   carve(sizeof(float) * HC);
  float*    pooled = (float*)   carve(sizeof(float) * N_GRAPH * HC);
  float*    cnt    = (float*)   carve(sizeof(float) * N_GRAPH);

  const int T = 256;
  auto cdiv = [](long a, long b) { return (int)((a + b - 1) / b); };

  // stage 0: self-loop attrs + augmented edge list
  gat_fill_f32<<<cdiv(N_NODES, T), T, 0, stream>>>(deg, 0.0f, N_NODES);
  gat_fill_f32<<<cdiv(N_NODES * DE, T), T, 0, stream>>>(esum, 0.0f, N_NODES * DE);
  gat_selfloop_accum<<<cdiv(N_EDGES, T), T, 0, stream>>>(ei, eattr, deg, esum);
  gat_build_edges<<<cdiv(E2, T), T, 0, stream>>>(ei, eattr, deg, esum, src2, dst2, ea2);

  auto gemm = [&](const float* A, const float* B, float* C, int M, int K, int Nc) {
    int blocks = cdiv(M, 128) * (Nc / 64);
    gat_gemm_wmma<<<blocks, 256, 0, stream>>>(A, B, C, M, K, Nc);
  };
  gemm(x, Wl1, xl, N_NODES, DIN, HC);
  gemm(x, Wr1, xr, N_NODES, DIN, HC);

  auto edge_layer = [&](const float* We, const float* att) {
    gat_fill_u32<<<cdiv(N_NODES * N_HEAD, T), T, 0, stream>>>(mkey, ENC_NEG_INF, N_NODES * N_HEAD);
    gat_fill_f32<<<cdiv(N_NODES * N_HEAD, T), T, 0, stream>>>(ssum, 0.0f, N_NODES * N_HEAD);
    gat_fill_f32<<<cdiv((long)N_NODES * HC, T), T, 0, stream>>>(agg, 0.0f, N_NODES * HC);
    const int ET = E2 * N_HEAD;
    gat_edge_score<<<cdiv(ET, T), T, 0, stream>>>(xl, xr, src2, dst2, ea2, We, att, wbuf, mkey);
    gat_edge_expsum<<<cdiv(ET, T), T, 0, stream>>>(dst2, mkey, wbuf, ssum);
    gat_edge_aggregate<<<cdiv(ET, T), T, 0, stream>>>(xl, src2, dst2, wbuf, ssum, agg);
  };

  // layer 1
  edge_layer(We1, att1);
  gat_bias_elu<<<cdiv((long)N_NODES * HC, T), T, 0, stream>>>(agg, bias1);
  gat_bn_stats<<<HC, 256, 0, stream>>>(agg, bn_mean, bn_rstd, N_NODES);
  gat_bn_apply<<<cdiv((long)N_NODES * HC, T), T, 0, stream>>>(agg, bn_mean, bn_rstd, gamma, beta, hbn);

  // layer 2
  gemm(hbn, Wl2, xl, N_NODES, HC, HC);
  gemm(hbn, Wr2, xr, N_NODES, HC, HC);
  edge_layer(We2, att2);

  // pool + head
  gat_fill_f32<<<cdiv(N_GRAPH * HC, T), T, 0, stream>>>(pooled, 0.0f, N_GRAPH * HC);
  gat_fill_f32<<<1, 32, 0, stream>>>(cnt, 0.0f, N_GRAPH);
  gat_pool<<<cdiv((long)N_NODES * HC, T), T, 0, stream>>>(agg, bias2, batch, pooled, cnt);
  gat_final<<<1, 512, 0, stream>>>(pooled, cnt, gamma, beta, Wlin, blin, out);
}
